// NodePPGN_3WL_79053168050212
// MI455X (gfx1250) — matrix operand order repaired and checked by reading, not compiled, not run on hardware
//
#include <hip/hip_runtime.h>

// ---------- problem constants ----------
// N=2048, M=256, C=256, HID=256 ; S/G are [256][512]

typedef __attribute__((ext_vector_type(16))) __bf16 v16bf;
typedef __attribute__((ext_vector_type(8)))  float  v8f;
typedef __attribute__((ext_vector_type(4)))  unsigned u32x4;
typedef __attribute__((ext_vector_type(8)))  int      i32x8;
typedef __attribute__((ext_vector_type(4)))  int      i32x4;

union FragU { v16bf v; uint4 q[2]; };

__device__ __forceinline__ unsigned short f2bf(float f) {
  unsigned u = __float_as_uint(f);
  u = u + 0x7FFFu + ((u >> 16) & 1u);            // round-to-nearest-even
  return (unsigned short)(u >> 16);
}
__device__ __forceinline__ unsigned pack2bf(float a, float b) {
  return (unsigned)f2bf(a) | ((unsigned)f2bf(b) << 16);
}

// A fragment, 16x32 bf16. lane<16: K0..7 (q0), K16..23 (q1); lane>=16: K8..15, K24..31.
__device__ __forceinline__ v16bf load_a(const __bf16* base, int stride, int row, int k0, int lane) {
  FragU f;
  const __bf16* p = base + row * stride + k0 + ((lane & 16) ? 8 : 0);
  f.q[0] = *(const uint4*)(p);
  f.q[1] = *(const uint4*)(p + 16);
  return f.v;
}
// B fragment, 32x16 bf16 from row-major [h][c]. lane<16: K0..7, K16..23; lane>=16: K8..15, K24..31.
__device__ __forceinline__ v16bf load_b(const __bf16* base, int stride, int row, int k0, int lane) {
  FragU f;
  const __bf16* p = base + row * stride + k0 + ((lane & 16) ? 8 : 0);
  f.q[0] = *(const uint4*)(p);
  f.q[1] = *(const uint4*)(p + 16);
  return f.v;
}

#if defined(__gfx1250__) && __has_builtin(__builtin_amdgcn_tensor_load_to_lds) && \
    __has_builtin(__builtin_amdgcn_s_wait_tensorcnt)
#define USE_TDM 1
#else
#define USE_TDM 0
#endif

#if USE_TDM
// LDS byte offset of a __shared__ object (generic -> LDS addrspace, then ptrtoint).
__device__ __forceinline__ unsigned lds_off(const void* p) {
  return (unsigned)(unsigned long long)(__attribute__((address_space(3))) const void*)p;
}
// TDM: load a 2D bf16 tile (rows x 256 elems, row stride 256 elems in memory) into LDS,
// hardware-padding 16B after every 512B row -> LDS row stride 264 bf16 (bank-conflict free).
__device__ __forceinline__ void tdm_load_w_tile(unsigned lds_addr, const void* gptr, int rows) {
  unsigned long long ga = (unsigned long long)(uintptr_t)gptr;
  u32x4 g0;
  g0[0] = 1u;                                           // count=1 (valid user descriptor)
  g0[1] = lds_addr;                                     // LDS byte address
  g0[2] = (unsigned)(ga & 0xFFFFFFFFu);                 // global_addr[31:0]
  g0[3] = (unsigned)((ga >> 32) & 0x01FFFFFFu) | 0x80000000u;  // global_addr[56:32] | type=2
  i32x8 g1;
  g1[0] = (int)((1u << 16) |                            // data_size = 2B
                (1u << 20) |                            // pad_enable
                (6u << 22) |                            // pad_interval: every 128 DWORDs (512B row)
                (3u << 25));                            // pad_amount: 4 DWORDs (16B)
  g1[1] = (int)(256u << 16);                            // tensor_dim0 = 256 (lo16 @ [31:16])
  g1[2] = (int)(256u << 16);                            // dim0 hi16=0 | tensor_dim1=256 (lo16)
  g1[3] = (int)(256u << 16);                            // dim1 hi16=0 | tile_dim0 = 256
  g1[4] = (int)(unsigned)rows;                          // tile_dim1 = rows | tile_dim2 = 0
  g1[5] = (int)256u;                                    // tensor_dim0_stride = 256 (lo32)
  g1[6] = 0;                                            // stride0 hi16 | stride1 lo16
  g1[7] = 0;
  i32x4 z4 = {0, 0, 0, 0};                              // groups 2/3 unused (2D tile)
  i32x8 z8 = {0, 0, 0, 0, 0, 0, 0, 0};                  // extra group (6-arg toolchain), unused
  __builtin_amdgcn_tensor_load_to_lds(g0, g1, z4, z4, z8, 0);
}
#endif

// ---------- pass 0: W1,W2 -> bf16 transposed [h][c] ----------
__global__ void k_pass0(const float* __restrict__ W1, const float* __restrict__ W2,
                        unsigned short* __restrict__ Wt1, unsigned short* __restrict__ Wt2) {
  int idx = blockIdx.x * blockDim.x + threadIdx.x;   // 65536
  if (idx < 65536) {
    int h = idx >> 8, c = idx & 255;
    Wt1[idx] = f2bf(W1[c * 256 + h]);
    Wt2[idx] = f2bf(W2[c * 256 + h]);
  }
}

// ---------- pass 1: per-chunk partial S ----------
// grid: (32 i-chunks of 64, 4 j-blocks of 64, 2 h-blocks of 128), 256 threads
__launch_bounds__(256)
__global__ void k_pass1(const float* __restrict__ X,
                        const float* __restrict__ b1, const float* __restrict__ b2,
                        const unsigned short* __restrict__ Wt1u, const unsigned short* __restrict__ Wt2u,
                        float* __restrict__ Spart) {
  __shared__ __align__(16) __bf16 Xs[64 * 264];   // 64 j rows x 256 c (+8 pad)
  __shared__ float Ss[64 * 260];                  // 64 j x (128 t1 | 128 t1*t2) (+4 pad)

  const int tid = threadIdx.x;
  const int lane = tid & 31, wave = tid >> 5;
  const int wj = wave & 1, wh = wave >> 1;        // 2 x 4 wave grid (32j x 32h per wave)
  const int lm = lane & 15;
  const int mhalf = (lane >> 4) << 3;             // 0 or 8
  const int chunk = blockIdx.x;                   // 0..31
  const int jb = blockIdx.y;                      // 0..3
  const int hb = blockIdx.z;                      // 0..1

  const __bf16* W1p = (const __bf16*)Wt1u;
  const __bf16* W2p = (const __bf16*)Wt2u;

  for (int p = tid; p < 64 * 260; p += 256) Ss[p] = 0.f;
  __syncthreads();

  const v8f zf = {0.f,0.f,0.f,0.f,0.f,0.f,0.f,0.f};

  for (int ii = 0; ii < 64; ++ii) {
    const int i = chunk * 64 + ii;
    // stage X[i, jb*64 .. +64, :] -> bf16 LDS
    const float* Xrow = X + ((size_t)i * 256 + jb * 64) * 256;
    for (int p = tid; p < 64 * 128; p += 256) {
      int r = p >> 7, cp = p & 127;
      const float2 xv = *(const float2*)(Xrow + r * 256 + 2 * cp);
      *(unsigned*)&Xs[r * 264 + 2 * cp] = pack2bf(xv.x, xv.y);
    }
    // pull the X tile of the next 64 j rows (next j-block) toward the caches while we compute (global_prefetch_b8)
    if (ii + 1 < 64)
      __builtin_prefetch(Xrow + 64 * 256 + tid * 64, 0, 1);
    __syncthreads();

    v8f t1a[2][2], t2a[2][2];
    #pragma unroll
    for (int tj = 0; tj < 2; ++tj)
      #pragma unroll
      for (int th = 0; th < 2; ++th) { t1a[tj][th] = zf; t2a[tj][th] = zf; }

    #pragma unroll
    for (int ks = 0; ks < 8; ++ks) {
      const int k0 = ks * 32;
      v16bf a0 = load_a(Xs, 264, wj * 32 + lm,      k0, lane);
      v16bf a1 = load_a(Xs, 264, wj * 32 + 16 + lm, k0, lane);
      #pragma unroll
      for (int th = 0; th < 2; ++th) {
        const int hrow = hb * 128 + wh * 32 + th * 16 + lm;    // global h row of Wt
        v16bf bB1 = load_b(W1p, 256, hrow, k0, lane);
        v16bf bB2 = load_b(W2p, 256, hrow, k0, lane);
        t1a[0][th] = __builtin_amdgcn_wmma_f32_16x16x32_bf16(false, a0, false, bB1, (short)0, t1a[0][th], false, false);
        t1a[1][th] = __builtin_amdgcn_wmma_f32_16x16x32_bf16(false, a1, false, bB1, (short)0, t1a[1][th], false, false);
        t2a[0][th] = __builtin_amdgcn_wmma_f32_16x16x32_bf16(false, a0, false, bB2, (short)0, t2a[0][th], false, false);
        t2a[1][th] = __builtin_amdgcn_wmma_f32_16x16x32_bf16(false, a1, false, bB2, (short)0, t2a[1][th], false, false);
      }
    }

    // accumulate H = [t1, t1*t2] into local S tile
    #pragma unroll
    for (int th = 0; th < 2; ++th) {
      const int hl = wh * 32 + th * 16 + lm;       // 0..127
      const int hg = hb * 128 + hl;                // 0..255
      const float bv1 = b1[hg], bv2 = b2[hg];
      #pragma unroll
      for (int tj = 0; tj < 2; ++tj) {
        #pragma unroll
        for (int r = 0; r < 8; ++r) {
          const int j = wj * 32 + tj * 16 + r + mhalf;  // 0..63
          const float t1 = t1a[tj][th][r] + bv1;
          const float t2 = t2a[tj][th][r] + bv2;
          Ss[j * 260 + hl]       += t1;
          Ss[j * 260 + 128 + hl] += t1 * t2;
        }
      }
    }
    __syncthreads();
  }

  // flush partial S tile (deterministic: disjoint regions per workgroup)
  float* Sp = Spart + ((size_t)chunk * 256 + jb * 64) * 512;
  for (int p = tid; p < 64 * 256; p += 256) {
    const int j = p >> 8, hl = p & 255;
    const int hg2 = (hl < 128) ? (hb * 128 + hl) : (256 + hb * 128 + (hl - 128));
    Sp[(size_t)j * 512 + hg2] = Ss[j * 260 + hl];
  }
}

// ---------- pass G: reduce chunk partials, fold Wc ----------
__global__ void k_passG(const float* __restrict__ Spart, const float* __restrict__ Wc,
                        float* __restrict__ G) {
  int idx = blockIdx.x * blockDim.x + threadIdx.x;   // 131072
  if (idx < 131072) {
    float s = 0.f;
    #pragma unroll 4
    for (int c = 0; c < 32; ++c) s += Spart[(size_t)c * 131072 + idx];
    G[idx] = s * Wc[idx & 511];
  }
}

// ---------- pass 2: logits partials ----------
// grid: (2048 i, 4 j-blocks of 64), 256 threads
__launch_bounds__(256)
__global__ void k_pass2(const float* __restrict__ X,
                        const float* __restrict__ b1, const float* __restrict__ b2,
                        const unsigned short* __restrict__ Wt1u, const unsigned short* __restrict__ Wt2u,
                        const float* __restrict__ G, float* __restrict__ P2) {
  __shared__ __align__(16) __bf16 Xs[64 * 264];
  __shared__ __align__(16) __bf16 W1s[64 * 264];
  __shared__ __align__(16) __bf16 W2s[64 * 264];
  __shared__ float red[256];

  const int tid = threadIdx.x;
  const int lane = tid & 31, wave = tid >> 5;
  const int wj = wave & 1, wh = wave >> 1;        // 2 x 4 waves (32j x 16h per wave)
  const int lm = lane & 15;
  const int mhalf = (lane >> 4) << 3;
  const int i  = blockIdx.x;                      // 0..2047
  const int jb = blockIdx.y;                      // 0..3

  // stage X tile once
  const float* Xrow = X + ((size_t)i * 256 + jb * 64) * 256;
  for (int p = tid; p < 64 * 128; p += 256) {
    int r = p >> 7, cp = p & 127;
    const float2 xv = *(const float2*)(Xrow + r * 256 + 2 * cp);
    *(unsigned*)&Xs[r * 264 + 2 * cp] = pack2bf(xv.x, xv.y);
  }

#if USE_TDM
  const unsigned w1s_addr = lds_off(W1s);
  const unsigned w2s_addr = lds_off(W2s);
#else
  const unsigned* W1u32 = (const unsigned*)Wt1u;
  const unsigned* W2u32 = (const unsigned*)Wt2u;
#endif

  const v8f zf = {0.f,0.f,0.f,0.f,0.f,0.f,0.f,0.f};
  float acc = 0.f;

  for (int hbk = 0; hbk < 4; ++hbk) {             // 64 h values per iteration
    __syncthreads();                               // X staged / previous W reads done
#if USE_TDM
    // Tensor Data Mover: DMA both W tiles into LDS with hardware row padding.
    if (wave == 0) {
      tdm_load_w_tile(w1s_addr, Wt1u + hbk * 64 * 256, 64);
      tdm_load_w_tile(w2s_addr, Wt2u + hbk * 64 * 256, 64);
      __builtin_amdgcn_s_wait_tensorcnt(0);
    }
#else
    for (int p = tid; p < 64 * 128; p += 256) {    // stage W tiles (bf16, uint copies)
      int r = p >> 7, cu = p & 127;
      *(unsigned*)&W1s[r * 264 + 2 * cu] = W1u32[(hbk * 64 + r) * 128 + cu];
      *(unsigned*)&W2s[r * 264 + 2 * cu] = W2u32[(hbk * 64 + r) * 128 + cu];
    }
#endif
    __syncthreads();

    v8f t1a[2], t2a[2];
    t1a[0] = zf; t1a[1] = zf; t2a[0] = zf; t2a[1] = zf;

    #pragma unroll
    for (int ks = 0; ks < 8; ++ks) {
      const int k0 = ks * 32;
      v16bf a0 = load_a(Xs, 264, wj * 32 + lm,      k0, lane);
      v16bf a1 = load_a(Xs, 264, wj * 32 + 16 + lm, k0, lane);
      const int hrow = wh * 16 + lm;               // local W row
      v16bf bB1 = load_b(W1s, 264, hrow, k0, lane);
      v16bf bB2 = load_b(W2s, 264, hrow, k0, lane);
      t1a[0] = __builtin_amdgcn_wmma_f32_16x16x32_bf16(false, a0, false, bB1, (short)0, t1a[0], false, false);
      t1a[1] = __builtin_amdgcn_wmma_f32_16x16x32_bf16(false, a1, false, bB1, (short)0, t1a[1], false, false);
      t2a[0] = __builtin_amdgcn_wmma_f32_16x16x32_bf16(false, a0, false, bB2, (short)0, t2a[0], false, false);
      t2a[1] = __builtin_amdgcn_wmma_f32_16x16x32_bf16(false, a1, false, bB2, (short)0, t2a[1], false, false);
    }

    // contract against G (L2-hot, 512KB)
    const int hg = hbk * 64 + wh * 16 + lm;        // 0..255
    const float bv1 = b1[hg], bv2 = b2[hg];
    #pragma unroll
    for (int tj = 0; tj < 2; ++tj) {
      #pragma unroll
      for (int r = 0; r < 8; ++r) {
        const int jg = jb * 64 + wj * 32 + tj * 16 + r + mhalf;  // 0..255
        const float t1 = t1a[tj][r] + bv1;
        const float t2 = t2a[tj][r] + bv2;
        acc += t1 * G[jg * 512 + hg] + (t1 * t2) * G[jg * 512 + 256 + hg];
      }
    }
  }

  red[tid] = acc;
  __syncthreads();
  #pragma unroll
  for (int s = 128; s > 0; s >>= 1) {
    if (tid < s) red[tid] += red[tid + s];
    __syncthreads();
  }
  if (tid == 0) P2[i * 4 + jb] = red[0];
}

// ---------- pass 3: final logits ----------
__global__ void k_pass3(const float* __restrict__ P2, const float* __restrict__ bc,
                        float* __restrict__ logits) {
  int i = blockIdx.x * blockDim.x + threadIdx.x;
  if (i < 2048)
    logits[i] = bc[0] + P2[i * 4 + 0] + P2[i * 4 + 1] + P2[i * 4 + 2] + P2[i * 4 + 3];
}

extern "C" void kernel_launch(void* const* d_in, const int* in_sizes, int n_in,
                              void* d_out, int out_size, void* d_ws, size_t ws_size,
                              hipStream_t stream) {
  const float* X  = (const float*)d_in[0];
  const float* W1 = (const float*)d_in[1];
  const float* b1 = (const float*)d_in[2];
  const float* W2 = (const float*)d_in[3];
  const float* b2 = (const float*)d_in[4];
  const float* Wc = (const float*)d_in[5];
  const float* bc = (const float*)d_in[6];
  float* logits = (float*)d_out;

  // workspace layout
  float* Spart = (float*)d_ws;                          // 32 * 256 * 512 f32 = 16 MB
  float* G     = Spart + (size_t)32 * 256 * 512;        // 256*512 f32 = 512 KB
  unsigned short* Wt1 = (unsigned short*)(G + 256 * 512);  // 128 KB
  unsigned short* Wt2 = Wt1 + 256 * 256;                   // 128 KB
  float* P2 = (float*)(Wt2 + 256 * 256);                   // 2048*4 f32 = 32 KB

  k_pass0<<<dim3(256), dim3(256), 0, stream>>>(W1, W2, Wt1, Wt2);
  k_pass1<<<dim3(32, 4, 2), dim3(256), 0, stream>>>(X, b1, b2, Wt1, Wt2, Spart);
  k_passG<<<dim3(512), dim3(256), 0, stream>>>(Spart, Wc, G);
  k_pass2<<<dim3(2048, 4), dim3(256), 0, stream>>>(X, b1, b2, Wt1, Wt2, G, P2);
  k_pass3<<<dim3(8), dim3(256), 0, stream>>>(P2, bc, logits);
}
